// GCN_62792421867777
// MI455X (gfx1250) — matrix operand order repaired and checked by reading, not compiled, not run on hardware
//
#include <hip/hip_runtime.h>
#include <hip/hip_bf16.h>

typedef __attribute__((ext_vector_type(16))) __bf16 v16bf;
typedef __attribute__((ext_vector_type(8)))  float  v8f;

#define HID 128
#define NGRAPH 64
#define OUTF 10

// ---------------------------------------------------------------------------
// Utility kernels
// ---------------------------------------------------------------------------
__global__ void fill_f32(float* __restrict__ p, float v, size_t n) {
    size_t i = (size_t)blockIdx.x * blockDim.x + threadIdx.x;
    if (i < n) p[i] = v;
}

__global__ void fill_i32(int* __restrict__ p, int v, size_t n) {
    size_t i = (size_t)blockIdx.x * blockDim.x + threadIdx.x;
    if (i < n) p[i] = v;
}

// In-degree count (int atomics, once per launch)
__global__ void edge_count(const int* __restrict__ dst, int* __restrict__ cnt,
                           int E, int N) {
    int e = blockIdx.x * blockDim.x + threadIdx.x;
    if (e >= E) return;
    int d = dst[e];
    d = min(max(d, 0), N - 1);
    atomicAdd(&cnt[d], 1);
}

// dis[i] = rsqrt(1 + in_degree)   (self loop included; always > 0)
__global__ void dis_from_count(const int* __restrict__ cnt,
                               float* __restrict__ dis, int N) {
    int i = blockIdx.x * blockDim.x + threadIdx.x;
    if (i < N) dis[i] = __frsqrt_rn(1.0f + (float)cnt[i]);
}

// Single-workgroup exclusive scan of cnt -> row_start (and cursor copy).
// 1024 threads, Hillis-Steele per 1024-chunk with running carry.
__global__ void __launch_bounds__(1024)
exscan(const int* __restrict__ cnt, int* __restrict__ row_start,
       int* __restrict__ cursor, int N) {
    __shared__ int lds[1024];
    __shared__ int carry;
    const int tid = threadIdx.x;
    if (tid == 0) carry = 0;
    __syncthreads();
    for (int base = 0; base < N; base += 1024) {
        const int i = base + tid;
        const int v = (i < N) ? cnt[i] : 0;
        lds[tid] = v;
        __syncthreads();
#pragma unroll
        for (int off = 1; off < 1024; off <<= 1) {
            const int t = (tid >= off) ? lds[tid - off] : 0;
            __syncthreads();
            lds[tid] += t;
            __syncthreads();
        }
        const int c = carry;            // uniform; stable since last barrier
        const int excl = c + lds[tid] - v;
        if (i < N) {
            row_start[i] = excl;
            cursor[i]    = excl;
        }
        __syncthreads();
        if (tid == 0) carry = c + lds[1023];
        __syncthreads();
    }
    if (tid == 0) row_start[N] = carry;
}

// Bucket-fill CSR: one int atomic per edge (once per launch)
__global__ void csr_fill(const int* __restrict__ src, const int* __restrict__ dst,
                         int* __restrict__ cursor, int* __restrict__ csr_src,
                         int E, int N) {
    int e = blockIdx.x * blockDim.x + threadIdx.x;
    if (e >= E) return;
    int s = src[e];
    int d = dst[e];
    s = min(max(s, 0), N - 1);
    d = min(max(d, 0), N - 1);
    const int pos = atomicAdd(&cursor[d], 1);
    csr_src[pos] = s;
}

// ---------------------------------------------------------------------------
// GEMM: C[M x 128] = A[M x 128] @ W[128 x 128], bf16 WMMA, f32 accumulate.
// One wave per 16-row strip; 8 waves (256 thr) per block; W^T in LDS as bf16.
// Optional fused bias + ReLU epilogue.
// ---------------------------------------------------------------------------
__global__ void __launch_bounds__(256)
gemm128_wmma_bf16(const float* __restrict__ A, const float* __restrict__ Wg,
                  const float* __restrict__ bias, float* __restrict__ C,
                  int M, int fuse_bias_relu) {
    __shared__ __bf16 Wt[HID * HID];   // transposed: Wt[n*128 + k], 32 KB

    // Cooperative load+transpose+convert of W (row-major [k][n] in global)
    for (int j = 0; j < (HID * HID) / 256; ++j) {
        int e = threadIdx.x + j * 256;
        int k = e >> 7;
        int n = e & 127;
        Wt[n * HID + k] = (__bf16)Wg[e];
    }
    __syncthreads();

    const int wave = threadIdx.x >> 5;
    const int lane = threadIdx.x & 31;
    const int tile = blockIdx.x * 8 + wave;
    const int row0 = tile * 16;
    if (row0 >= M) return;                 // wave-uniform

    const int halfsel = (lane >> 4) & 1;   // 0: lanes 0-15, 1: lanes 16-31
    int arow = row0 + (lane & 15);
    if (arow > M - 1) arow = M - 1;        // clamp (guarded on store)
    const float4* __restrict__ Arow4 =
        reinterpret_cast<const float4*>(A + (size_t)arow * HID);

    const int koffA = halfsel * 16;        // A fragment K offset within chunk
    const int koffB = halfsel * 16;        // B fragment K offset within chunk

    v8f acc[8];
#pragma unroll
    for (int nt = 0; nt < 8; ++nt) acc[nt] = (v8f){0, 0, 0, 0, 0, 0, 0, 0};

#pragma unroll
    for (int kc = 0; kc < 4; ++kc) {
        const int k0 = kc * 32;
        // A fragment: lane holds row (lane&15); K runs {k0+koffA..+15}
        union { v16bf v; __bf16 e[16]; float4 _dummy; } af;
        const float4 p0 = Arow4[(k0 + koffA) >> 2];
        const float4 p1 = Arow4[((k0 + koffA) >> 2) + 1];
        const float4 p2 = Arow4[((k0 + koffA) >> 2) + 2];
        const float4 p3 = Arow4[((k0 + koffA) >> 2) + 3];
        af.e[0]  = (__bf16)p0.x; af.e[1]  = (__bf16)p0.y;
        af.e[2]  = (__bf16)p0.z; af.e[3]  = (__bf16)p0.w;
        af.e[4]  = (__bf16)p1.x; af.e[5]  = (__bf16)p1.y;
        af.e[6]  = (__bf16)p1.z; af.e[7]  = (__bf16)p1.w;
        af.e[8]  = (__bf16)p2.x; af.e[9]  = (__bf16)p2.y;
        af.e[10] = (__bf16)p2.z; af.e[11] = (__bf16)p2.w;
        af.e[12] = (__bf16)p3.x; af.e[13] = (__bf16)p3.y;
        af.e[14] = (__bf16)p3.z; af.e[15] = (__bf16)p3.w;

#pragma unroll
        for (int nt = 0; nt < 8; ++nt) {
            // B fragment: lane holds col N = nt*16 + (lane&15); 16 contiguous K
            const v16bf bf = *reinterpret_cast<const v16bf*>(
                &Wt[(nt * 16 + (lane & 15)) * HID + k0 + koffB]);
            acc[nt] = __builtin_amdgcn_wmma_f32_16x16x32_bf16(
                false, af.v, false, bf, (short)0, acc[nt], false, false);
        }
    }

    // Epilogue: C layout VGPR r -> M = r + 8*halfsel, N = lane&15 within tile
    const int col0 = lane & 15;
#pragma unroll
    for (int nt = 0; nt < 8; ++nt) {
        const int n = nt * 16 + col0;
        const float bv = fuse_bias_relu ? bias[n] : 0.0f;
#pragma unroll
        for (int r = 0; r < 8; ++r) {
            const int row = row0 + r + halfsel * 8;
            if (row < M) {
                float v = acc[nt][r];
                if (fuse_bias_relu) v = fmaxf(v + bv, 0.0f);
                C[(size_t)row * HID + n] = v;
            }
        }
    }
}

// ---------------------------------------------------------------------------
// Fused GCN aggregation: h[d] = relu( sum_{s in N(d)} t[s]*dis[s]*dis[d]
//                                     + t[d]*dis[d]^2 + bias )
// One wave per destination node; lane owns one float4 of the 128-wide row.
// Unroll-by-4 over incoming edges for memory-level parallelism.
// ---------------------------------------------------------------------------
__global__ void __launch_bounds__(256)
gcn_aggregate(const float* __restrict__ t, const int* __restrict__ row_start,
              const int* __restrict__ csr_src, const float* __restrict__ dis,
              const float* __restrict__ bias, float* __restrict__ h, int N) {
    const int node = blockIdx.x * (blockDim.x >> 5) + (threadIdx.x >> 5);
    if (node >= N) return;                 // wave-uniform
    const int lane = threadIdx.x & 31;
    const int foff = lane * 4;

    const float dd  = dis[node];
    const int   beg = row_start[node];
    const int   end = row_start[node + 1];

    float4 acc = {0.0f, 0.0f, 0.0f, 0.0f};
    int j = beg;
    // 4-wide software pipeline: batch index loads, dis loads, feature gathers
    for (; j + 3 < end; j += 4) {
        const int s0 = csr_src[j + 0];
        const int s1 = csr_src[j + 1];
        const int s2 = csr_src[j + 2];
        const int s3 = csr_src[j + 3];
        const float n0 = dis[s0] * dd;
        const float n1 = dis[s1] * dd;
        const float n2 = dis[s2] * dd;
        const float n3 = dis[s3] * dd;
        const float4 v0 = *reinterpret_cast<const float4*>(&t[(size_t)s0 * HID + foff]);
        const float4 v1 = *reinterpret_cast<const float4*>(&t[(size_t)s1 * HID + foff]);
        const float4 v2 = *reinterpret_cast<const float4*>(&t[(size_t)s2 * HID + foff]);
        const float4 v3 = *reinterpret_cast<const float4*>(&t[(size_t)s3 * HID + foff]);
        acc.x += v0.x * n0; acc.y += v0.y * n0; acc.z += v0.z * n0; acc.w += v0.w * n0;
        acc.x += v1.x * n1; acc.y += v1.y * n1; acc.z += v1.z * n1; acc.w += v1.w * n1;
        acc.x += v2.x * n2; acc.y += v2.y * n2; acc.z += v2.z * n2; acc.w += v2.w * n2;
        acc.x += v3.x * n3; acc.y += v3.y * n3; acc.z += v3.z * n3; acc.w += v3.w * n3;
    }
    for (; j < end; ++j) {
        const int s = csr_src[j];
        const float nrm = dis[s] * dd;
        const float4 v = *reinterpret_cast<const float4*>(&t[(size_t)s * HID + foff]);
        acc.x += v.x * nrm; acc.y += v.y * nrm;
        acc.z += v.z * nrm; acc.w += v.w * nrm;
    }
    // self loop (norm = dis[d]^2) + bias + ReLU, fused
    const float sl = dd * dd;
    const float4 vs = *reinterpret_cast<const float4*>(&t[(size_t)node * HID + foff]);
    const float4 bv = *reinterpret_cast<const float4*>(&bias[foff]);
    float4 o;
    o.x = fmaxf(acc.x + vs.x * sl + bv.x, 0.0f);
    o.y = fmaxf(acc.y + vs.y * sl + bv.y, 0.0f);
    o.z = fmaxf(acc.z + vs.z * sl + bv.z, 0.0f);
    o.w = fmaxf(acc.w + vs.w * sl + bv.w, 0.0f);
    *reinterpret_cast<float4*>(&h[(size_t)node * HID + foff]) = o;
}

// ---------------------------------------------------------------------------
// Pool: batch ids are sorted, so accumulate runs in registers and flush one
// atomic per (graph-boundary, feature).  Block = 128 threads, 64 nodes/block.
// ---------------------------------------------------------------------------
#define POOL_SPAN 64
__global__ void __launch_bounds__(128)
pool_sum(const float* __restrict__ h, const int* __restrict__ batch,
         float* __restrict__ sums, float* __restrict__ cnts, int N) {
    const int base = blockIdx.x * POOL_SPAN;
    if (base >= N) return;
    const int f = threadIdx.x;
    float acc = 0.0f;
    float cnt = 0.0f;
    int curg = -1;
    for (int k = 0; k < POOL_SPAN; ++k) {
        const int i = base + k;
        if (i >= N) break;
        int g = batch[i];
        g = min(max(g, 0), NGRAPH - 1);
        if (g != curg) {
            if (curg >= 0) {
                atomicAdd(&sums[curg * HID + f], acc);
                if (f == 0) atomicAdd(&cnts[curg], cnt);
            }
            acc = 0.0f;
            cnt = 0.0f;
            curg = g;
        }
        acc += h[(size_t)i * HID + f];
        cnt += 1.0f;
    }
    if (curg >= 0) {
        atomicAdd(&sums[curg * HID + f], acc);
        if (f == 0) atomicAdd(&cnts[curg], cnt);
    }
}

// ---------------------------------------------------------------------------
// Classifier: out[g,o] = (sums[g]/max(cnt,1)) @ W_cls + b_cls
// ---------------------------------------------------------------------------
__global__ void __launch_bounds__(640)
classifier(const float* __restrict__ sums, const float* __restrict__ cnts,
           const float* __restrict__ Wc, const float* __restrict__ bc,
           float* __restrict__ out) {
    const int tid = threadIdx.x;
    if (tid >= NGRAPH * OUTF) return;
    const int g = tid / OUTF;
    const int o = tid % OUTF;
    const float inv = 1.0f / fmaxf(cnts[g], 1.0f);
    float acc = bc[o];
#pragma unroll 4
    for (int k = 0; k < HID; ++k)
        acc += (sums[g * HID + k] * inv) * Wc[k * OUTF + o];
    out[g * OUTF + o] = acc;
}

// ---------------------------------------------------------------------------
// Host launcher
// ---------------------------------------------------------------------------
extern "C" void kernel_launch(void* const* d_in, const int* in_sizes, int n_in,
                              void* d_out, int out_size, void* d_ws, size_t ws_size,
                              hipStream_t stream) {
    const float* x     = (const float*)d_in[0];
    const int*   ei    = (const int*)d_in[1];
    const int*   batch = (const int*)d_in[2];
    const float* W_pre = (const float*)d_in[3];
    const float* b_pre = (const float*)d_in[4];
    const float* Ws[3] = {(const float*)d_in[5], (const float*)d_in[7],
                          (const float*)d_in[9]};
    const float* bs[3] = {(const float*)d_in[6], (const float*)d_in[8],
                          (const float*)d_in[10]};
    const float* W_cls = (const float*)d_in[11];
    const float* b_cls = (const float*)d_in[12];
    float* out = (float*)d_out;

    const int N = in_sizes[0] / HID;
    const int E = in_sizes[1] / 2;
    const int* src = ei;
    const int* dst = ei + E;

    // Workspace layout: dis(f32,N) | h(f32,N*128) | t(f32,N*128) |
    //                   cnt(i32,N) | row_start(i32,N+1) | cursor(i32,N) |
    //                   csr_src(i32,E) | sums(f32,64*128) | cnts(f32,64)
    const size_t nF   = (size_t)N * HID;
    const size_t Npad = ((size_t)N + 255) & ~(size_t)255;
    float* dis       = (float*)d_ws;
    float* h         = dis + Npad;
    float* t         = h + nF;
    int*   cnt       = (int*)(t + nF);
    int*   row_start = cnt + Npad;
    int*   cursor    = row_start + Npad;   // (>= N+1 slots available)
    int*   csr_src   = cursor + Npad;
    float* sums      = (float*)(csr_src + (((size_t)E + 255) & ~(size_t)255));
    float* cnts      = sums + NGRAPH * HID;

    const int thr = 256;
    const int nodeBlocks = (N + thr - 1) / thr;
    const int edgeBlocks = (E + thr - 1) / thr;
    const int gemmBlocks = (((N + 15) / 16) + 7) / 8;
    const int waveBlocks = (N + 7) / 8;    // 8 waves per 256-thread block

    // 1) build CSR-by-dst + symmetric normalization (once per launch)
    fill_i32<<<nodeBlocks, thr, 0, stream>>>(cnt, 0, (size_t)N);
    edge_count<<<edgeBlocks, thr, 0, stream>>>(dst, cnt, E, N);
    dis_from_count<<<nodeBlocks, thr, 0, stream>>>(cnt, dis, N);
    exscan<<<1, 1024, 0, stream>>>(cnt, row_start, cursor, N);
    csr_fill<<<edgeBlocks, thr, 0, stream>>>(src, dst, cursor, csr_src, E, N);

    // 2) pre-layer: h = relu(x @ W_pre + b_pre)
    gemm128_wmma_bf16<<<gemmBlocks, thr, 0, stream>>>(x, W_pre, b_pre, h, N, 1);

    // 3) three GCN conv layers: t = h@W, then fused gather/self/bias/relu -> h
    for (int l = 0; l < 3; ++l) {
        gemm128_wmma_bf16<<<gemmBlocks, thr, 0, stream>>>(h, Ws[l], nullptr, t,
                                                          N, 0);
        gcn_aggregate<<<waveBlocks, thr, 0, stream>>>(t, row_start, csr_src,
                                                      dis, bs[l], h, N);
    }

    // 4) global mean pool + classifier
    fill_f32<<<(NGRAPH * HID + NGRAPH + thr - 1) / thr, thr, 0, stream>>>(
        sums, 0.0f, (size_t)(NGRAPH * HID + NGRAPH));
    pool_sum<<<(N + POOL_SPAN - 1) / POOL_SPAN, HID, 0, stream>>>(
        h, batch, sums, cnts, N);
    classifier<<<1, NGRAPH * OUTF, 0, stream>>>(sums, cnts, W_cls, b_cls, out);
}
